// GCN_11759620456737
// MI455X (gfx1250) — matrix operand order repaired and checked by reading, not compiled, not run on hardware
//
#include <hip/hip_runtime.h>
#include <hip/hip_bf16.h>

typedef __bf16 bf16_t;
typedef __attribute__((ext_vector_type(16))) __bf16 v16bf;
typedef __attribute__((ext_vector_type(8)))  __bf16 v8bf;
typedef __attribute__((ext_vector_type(8)))  float  v8f;

#define BATCH 8
#define CH    256
#define NPTS  4096
#define KNN   16
#define BPAD  264   // padded LDS row stride (bf16 elems): 528 B, 16B-aligned rows

#define SHUF16(lo, hi) __builtin_shufflevector(lo, hi, 0,1,2,3,4,5,6,7,8,9,10,11,12,13,14,15)

// ---------------------------------------------------------------------------
// Kernel A: transpose x (B,C,N) f32 -> xtb (B,N,C) bf16, and sq[b,n] = ||x_n||^2
// ---------------------------------------------------------------------------
__global__ __launch_bounds__(256) void kxpose(const float* __restrict__ x,
                                              bf16_t* __restrict__ xtb,
                                              float* __restrict__ sq) {
    const int n = blockIdx.x * 256 + threadIdx.x;
    const int b = blockIdx.y;
    const float* xb = x + (size_t)b * CH * NPTS;
    bf16_t* row = xtb + ((size_t)b * NPTS + n) * CH;
    float acc = 0.f;
    for (int c = 0; c < CH; ++c) {
        float v = xb[(size_t)c * NPTS + n];
        acc += v * v;
        row[c] = (bf16_t)v;
    }
    sq[b * NPTS + n] = acc;
}

// ---------------------------------------------------------------------------
// Kernel W: split W (256 x 512) into W1 = W[:, :256], W2 = W[:, 256:], bf16.
// ---------------------------------------------------------------------------
__global__ __launch_bounds__(256) void kwconv(const float* __restrict__ W,
                                              bf16_t* __restrict__ W1,
                                              bf16_t* __restrict__ W2) {
    const int e = blockIdx.x * 256 + threadIdx.x;
    const int o = e >> 9;
    const int c = e & 511;
    const float v = W[e];
    if (c < 256) W1[o * 256 + c] = (bf16_t)v;
    else         W2[o * 256 + (c - 256)] = (bf16_t)v;
}

// ---------------------------------------------------------------------------
// Kernel B: fused Gram-matrix (WMMA bf16) + per-row top-16 selection.
// B-panels staged to LDS by async-DMA (double buffered); WMMA consumes LDS
// through a 1-deep register rotation; top-k scan parallelized 4x across all
// waves with a one-time merge. Scan loops kept rolled to limit I-cache bloat;
// insertion lists stay fully unrolled (register resident).
// score[i][j] = 2*dot(x_i,x_j) - sq[j]  (row-constant sq[i] dropped).
// ---------------------------------------------------------------------------
__global__ __launch_bounds__(256) void kgram_topk(const bf16_t* __restrict__ xtb,
                                                  const float* __restrict__ sq,
                                                  int* __restrict__ idxo) {
    __shared__ bf16_t bufB[2][64][BPAD];   // 2 x 64 rows x 256 feats (+pad)
    __shared__ float  sc[64][65];
    __shared__ float  ssq[64];
    __shared__ float  mv[64][4][16];       // partial top-k merge scratch
    __shared__ int    mi[64][4][16];

    const int tid  = threadIdx.x;
    const int lane = tid & 31;
    const int w    = tid >> 5;
    const int b    = blockIdx.y;
    const int i0   = blockIdx.x * 64;
    const bf16_t* xb = xtb + (size_t)b * NPTS * CH;

    const int rt   = w & 3;        // row-tile of 16 rows
    const int cset = w >> 2;       // which 32-column half

    // ---- hoist all 8 A fragments (K=256) into registers -------------------
    const bf16_t* arow =
        xb + (size_t)(i0 + rt * 16 + (lane & 15)) * CH + ((lane >> 4) << 3);
    v16bf afr[8];
#pragma unroll
    for (int ks = 0; ks < 8; ++ks) {
        v8bf alo = *reinterpret_cast<const v8bf*>(arow + ks * 32);
        v8bf ahi = *reinterpret_cast<const v8bf*>(arow + ks * 32 + 16);
        afr[ks] = SHUF16(alo, ahi);
    }

    // per-thread partial top-16: row = tid&63, column quarter = tid>>6
    const int myrow = tid & 63;
    const int myq   = tid >> 6;
    float tv[16]; int ti[16];
#pragma unroll
    for (int q = 0; q < 16; ++q) { tv[q] = -3.0e38f; ti[q] = 0; }

    // async DMA: wave w copies rows {w, w+8, ..., w+56}; one instr = 1 row.
    auto issue_panel = [&](int j0, int nb) {
#pragma unroll
        for (int i = 0; i < 8; ++i) {
            const int r = w + 8 * i;
            unsigned int ldsa =
                (unsigned int)(size_t)&bufB[nb][r][lane * 8];
            unsigned long long ga =
                (unsigned long long)(size_t)(xb + (size_t)(j0 + r) * CH + lane * 8);
            asm volatile("global_load_async_to_lds_b128 %0, %1, off"
                         :: "v"(ldsa), "v"(ga) : "memory");
        }
    };

    issue_panel(0, 0);

    for (int j0 = 0; j0 < NPTS; j0 += 64) {
        const int cur = (j0 >> 6) & 1;
        if (j0 + 64 < NPTS) {
            issue_panel(j0 + 64, cur ^ 1);    // prefetch next panel
            asm volatile("s_wait_asynccnt 0x8" ::: "memory"); // panel `cur` landed
        } else {
            asm volatile("s_wait_asynccnt 0x0" ::: "memory");
        }
        __syncthreads();   // panel visible; prev scan done with sc

        // ---- WMMA phase: two 16x16 tiles per wave, B from LDS, 1-deep pipe
        const bf16_t* b0p = &bufB[cur][cset * 32 + (lane & 15)][(lane >> 4) << 3];
        const bf16_t* b1p = b0p + 16 * BPAD;
        v8f acc0 = {}, acc1 = {};

        v16bf b0c, b1c, b0n, b1n;
        {
            v8bf l0 = *reinterpret_cast<const v8bf*>(b0p);
            v8bf h0 = *reinterpret_cast<const v8bf*>(b0p + 16);
            v8bf l1 = *reinterpret_cast<const v8bf*>(b1p);
            v8bf h1 = *reinterpret_cast<const v8bf*>(b1p + 16);
            b0c = SHUF16(l0, h0);
            b1c = SHUF16(l1, h1);
        }
#pragma unroll
        for (int ks = 0; ks < 8; ++ks) {
            if (ks < 7) {   // prefetch next fragments BEFORE consuming current
                v8bf l0 = *reinterpret_cast<const v8bf*>(b0p + (ks + 1) * 32);
                v8bf h0 = *reinterpret_cast<const v8bf*>(b0p + (ks + 1) * 32 + 16);
                v8bf l1 = *reinterpret_cast<const v8bf*>(b1p + (ks + 1) * 32);
                v8bf h1 = *reinterpret_cast<const v8bf*>(b1p + (ks + 1) * 32 + 16);
                b0n = SHUF16(l0, h0);
                b1n = SHUF16(l1, h1);
            } else {
                b0n = b0c; b1n = b1c;
            }
            acc0 = __builtin_amdgcn_wmma_f32_16x16x32_bf16(
                       false, afr[ks], false, b0c, (short)0, acc0, false, false);
            acc1 = __builtin_amdgcn_wmma_f32_16x16x32_bf16(
                       false, afr[ks], false, b1c, (short)0, acc1, false, false);
            b0c = b0n; b1c = b1n;
        }

        // sc writes are already separated from prev scan by the top barrier
        const int rrow = rt * 16 + ((lane >> 4) << 3);
        const int col0 = cset * 32 + (lane & 15);
#pragma unroll
        for (int r = 0; r < 8; ++r) {
            sc[rrow + r][col0]      = acc0[r];
            sc[rrow + r][col0 + 16] = acc1[r];
        }
        if (tid < 64) ssq[tid] = sq[b * NPTS + j0 + tid];
        __syncthreads();

        // ---- top-k phase: all 256 threads; thread owns (row, 16-col quarter)
        {
            const int jb = myq * 16;
#pragma unroll 1
            for (int jj = 0; jj < 16; ++jj) {
                const float s = 2.f * sc[myrow][jb + jj] - ssq[jb + jj];
                if (s > tv[15]) {
                    float nv = s; int nj = j0 + jb + jj;
#pragma unroll
                    for (int q = 0; q < 16; ++q) {
                        if (nv > tv[q]) {
                            float ov = tv[q]; int oj = ti[q];
                            tv[q] = nv; ti[q] = nj; nv = ov; nj = oj;
                        }
                    }
                }
            }
        }
    }

    // ---- final merge: 4 partial lists per row -> global top-16 ------------
#pragma unroll
    for (int e = 0; e < 16; ++e) { mv[myrow][myq][e] = tv[e]; mi[myrow][myq][e] = ti[e]; }
    __syncthreads();

    if (tid < 64) {
        float fv[16]; int fi[16];
#pragma unroll
        for (int q = 0; q < 16; ++q) { fv[q] = -3.0e38f; fi[q] = 0; }
#pragma unroll 1
        for (int q = 0; q < 4; ++q) {
#pragma unroll 1
            for (int e = 0; e < 16; ++e) {
                const float s = mv[tid][q][e];
                if (s > fv[15]) {
                    float nv = s; int nj = mi[tid][q][e];
#pragma unroll
                    for (int u = 0; u < 16; ++u) {
                        if (nv > fv[u]) {
                            float ov = fv[u]; int oj = fi[u];
                            fv[u] = nv; fi[u] = nj; nv = ov; nj = oj;
                        }
                    }
                }
            }
        }
        int* op = idxo + ((size_t)b * NPTS + i0 + tid) * KNN;
#pragma unroll
        for (int q = 0; q < 16; ++q) op[q] = fi[q];
    }
}

// ---------------------------------------------------------------------------
// Kernel C: edge conv. t = W1*(nb_k - c) (WMMA, 16x256 tile per point),
// s = W2*c + bias (VALU), out[b,o,n] = max(0, max_k t[k][o] + s[o]).
// ---------------------------------------------------------------------------
__global__ __launch_bounds__(256) void kedge(const bf16_t* __restrict__ xtb,
                                             const int* __restrict__ idxi,
                                             const bf16_t* __restrict__ W1,
                                             const bf16_t* __restrict__ W2,
                                             const float* __restrict__ bias,
                                             float* __restrict__ out) {
    __shared__ float sS[8][256];

    const int tid  = threadIdx.x;
    const int lane = tid & 31;
    const int w    = tid >> 5;
    const int p    = blockIdx.x * 8 + w;     // flat point id = b*NPTS + n
    const int b    = p / NPTS;
    const int n    = p % NPTS;

    const bf16_t* cptr = xtb + (size_t)p * CH;
    const int k    = lane & 15;
    const int nidx = idxi[(size_t)p * KNN + k];
    const bf16_t* gptr = xtb + ((size_t)b * NPTS + nidx) * CH + ((lane >> 4) << 3);
    const bf16_t* cpl  = cptr + ((lane >> 4) << 3);

    // A fragments: (nb - center), 16(M=k) x 256(K=c) bf16
    v16bf afr[8];
#pragma unroll
    for (int ks = 0; ks < 8; ++ks) {
        const int c0 = ks * 32;
        v8bf nlo = *reinterpret_cast<const v8bf*>(gptr + c0);
        v8bf nhi = *reinterpret_cast<const v8bf*>(gptr + c0 + 16);
        v8bf clo = *reinterpret_cast<const v8bf*>(cpl + c0);
        v8bf chi = *reinterpret_cast<const v8bf*>(cpl + c0 + 16);
        v8bf dlo, dhi;
#pragma unroll
        for (int e = 0; e < 8; ++e) {
            dlo[e] = (bf16_t)((float)nlo[e] - (float)clo[e]);
            dhi[e] = (bf16_t)((float)nhi[e] - (float)chi[e]);
        }
        afr[ks] = SHUF16(dlo, dhi);
    }

    // k-independent term, vectorized: s[o] = bias[o] + sum_c center[c]*W2[o][c]
#pragma unroll
    for (int i = 0; i < 8; ++i) {
        const int o = lane + 32 * i;
        const bf16_t* wrow = W2 + (size_t)o * CH;
        float s = bias[o];
#pragma unroll 1
        for (int c8 = 0; c8 < CH; c8 += 8) {
            v8bf cv = *reinterpret_cast<const v8bf*>(cptr + c8);
            v8bf wv = *reinterpret_cast<const v8bf*>(wrow + c8);
#pragma unroll
            for (int e = 0; e < 8; ++e)
                s += (float)cv[e] * (float)wv[e];
        }
        sS[w][o] = s;
    }
    __syncthreads();

    float* outp = out + (size_t)b * CH * NPTS + n;
    const int koff = (lane >> 4) << 3;
#pragma unroll 1
    for (int ct = 0; ct < 16; ++ct) {
        const int o = ct * 16 + (lane & 15);
        const bf16_t* brow = W1 + (size_t)o * CH + koff;

        // batch-load all 8 B fragments (one clause), then 8 back-to-back WMMAs
        v16bf wf[8];
#pragma unroll
        for (int ks = 0; ks < 8; ++ks) {
            v8bf lo = *reinterpret_cast<const v8bf*>(brow + ks * 32);
            v8bf hi = *reinterpret_cast<const v8bf*>(brow + ks * 32 + 16);
            wf[ks] = SHUF16(lo, hi);
        }
        v8f acc = {};
#pragma unroll
        for (int ks = 0; ks < 8; ++ks)
            acc = __builtin_amdgcn_wmma_f32_16x16x32_bf16(
                      false, afr[ks], false, wf[ks], (short)0, acc, false, false);

        float m = acc[0];
#pragma unroll
        for (int r = 1; r < 8; ++r) m = fmaxf(m, acc[r]);
        m = fmaxf(m, __shfl_xor(m, 16, 32));
        const float res = fmaxf(m + sS[w][o], 0.f);
        if (lane < 16) outp[(size_t)o * NPTS] = res;
    }
}

// ---------------------------------------------------------------------------
extern "C" void kernel_launch(void* const* d_in, const int* in_sizes, int n_in,
                              void* d_out, int out_size, void* d_ws, size_t ws_size,
                              hipStream_t stream) {
    const float* x    = (const float*)d_in[0];   // (8, 256, 4096) f32
    const float* W    = (const float*)d_in[1];   // (256, 512) f32
    const float* bias = (const float*)d_in[2];   // (256,) f32
    float* out = (float*)d_out;                  // (8, 256, 4096) f32

    char* ws = (char*)d_ws;
    bf16_t* xtb = (bf16_t*)(ws);                      // 16,777,216 B : (B,N,C) bf16
    float*  sq  = (float*)(ws + 16777216);            //    131,072 B : (B,N) f32
    int*    idx = (int*)  (ws + 16908288);            //  2,097,152 B : (B,N,16) i32
    bf16_t* W1  = (bf16_t*)(ws + 19005440);           //    131,072 B
    bf16_t* W2  = (bf16_t*)(ws + 19136512);           //    131,072 B

    kxpose    <<<dim3(NPTS / 256, BATCH), 256, 0, stream>>>(x, xtb, sq);
    kwconv    <<<dim3((CH * 2 * CH) / 256), 256, 0, stream>>>(W, W1, W2);
    kgram_topk<<<dim3(NPTS / 64, BATCH), 256, 0, stream>>>(xtb, sq, idx);
    kedge     <<<dim3(BATCH * NPTS / 8), 256, 0, stream>>>(xtb, idx, W1, W2, bias, out);
}
